// SelfAttention_5274219839864
// MI455X (gfx1250) — matrix operand order repaired and checked by reading, not compiled, not run on hardware
//
#include <hip/hip_runtime.h>

// Self-attention for MI455X (gfx1250, wave32, WMMA, async global->LDS).
// Pipeline: cvt->bf16, QKV GEMMs (WMMA bf16 16x16x32), S=Q@K^T, row softmax,
// out = P@V. Intermediates live in d_ws and stay L2-resident (192MB L2).
// GEMM main loop uses GLOBAL_LOAD_ASYNC_TO_LDS_B128 (ASYNCcnt) with
// double-buffered LDS so fills overlap WMMA compute.

#define DIMD 1024
#define NTOK 4096

typedef __attribute__((ext_vector_type(16))) __bf16 v16bf;
typedef __attribute__((ext_vector_type(8)))  float  v8f;

static __device__ __forceinline__ unsigned short f32_to_bf16(float f) {
    unsigned int u = __float_as_uint(f);
    u += 0x7FFFu + ((u >> 16) & 1u);   // round-to-nearest-even
    return (unsigned short)(u >> 16);
}

// Direct memory->LDS async copy, 16B per lane. VDST = LDS byte address
// (low 32 bits of the generic pointer = wave-relative LDS offset, ISA 10.2).
static __device__ __forceinline__ void async_b128(unsigned lds_off, const void* gaddr) {
    asm volatile("global_load_async_to_lds_b128 %0, %1, off"
                 :: "v"(lds_off), "v"((unsigned long long)(size_t)gaddr)
                 : "memory");
}
static __device__ __forceinline__ void wait_async0() {
    asm volatile("s_wait_asynccnt 0x0" ::: "memory");
}

// ---------------------------------------------------------------- conversion
__global__ void __launch_bounds__(256) cvt_f32_bf16(const float* __restrict__ in,
                                                    unsigned short* __restrict__ out,
                                                    int n4) {
    int i = blockIdx.x * blockDim.x + threadIdx.x;
    if (i < n4) {
        float4 f = ((const float4*)in)[i];
        unsigned short h[4] = { f32_to_bf16(f.x), f32_to_bf16(f.y),
                                f32_to_bf16(f.z), f32_to_bf16(f.w) };
        ((uint2*)out)[i] = *(const uint2*)h;
    }
}

// ---------------------------------------------------------------- WMMA GEMM
// C[M,Ncols] = (A[M,K] @ B[Ncols,K]^T + bias) * scale
// A, B bf16 row-major (K contiguous). 256 threads = 8 wave32.
// Block tile 128x128, wave tile 32x64, KTILE=64, double-buffered async LDS.
// OUTMODE: 0 = fp32 row-major, 1 = bf16 row-major, 2 = bf16 transposed.
template <int OUTMODE>
__global__ void __launch_bounds__(256) wmma_gemm_abT(
    const unsigned short* __restrict__ A,
    const unsigned short* __restrict__ B,
    const float* __restrict__ bias, float scale,
    void* __restrict__ Cout, int ldc,
    int M, int Ncols, int K, int lda, int ldb)
{
    __shared__ unsigned short As[2][128 * 64];   // 2 x 16 KB
    __shared__ unsigned short Bs[2][128 * 64];   // 2 x 16 KB

    const int tid     = threadIdx.x;
    const int lane    = tid & 31;
    const int wave    = tid >> 5;
    const int wm      = wave & 3;        // 4 waves along M
    const int wn      = wave >> 2;       // 2 waves along N
    const int halfsel = lane >> 4;       // 0 => lanes 0-15, 1 => lanes 16-31
    const int l16     = lane & 15;

    const int m0 = blockIdx.y * 128;
    const int n0 = blockIdx.x * 128;

    v8f acc[2][4] = {};                  // 16x16 fp32 tiles, 8 VGPRs each

    union Frag { v16bf v; uint4 u[2]; };

    // Stage one 128x64 bf16 tile of A and B into LDS buffer `buf`
    // (4 async b128 per thread per matrix; 1024 vectors of 16B per tile).
    auto stage = [&](int buf, int k0) {
        #pragma unroll
        for (int j = 0; j < 4; ++j) {
            int id = tid + j * 256;          // 0..1023
            int r  = id >> 3;                // row 0..127
            int c  = (id & 7) * 8;           // bf16 col 0..56
            async_b128((unsigned)(size_t)(void*)&As[buf][r * 64 + c],
                       &A[(size_t)(m0 + r) * lda + k0 + c]);
            async_b128((unsigned)(size_t)(void*)&Bs[buf][r * 64 + c],
                       &B[(size_t)(n0 + r) * ldb + k0 + c]);
        }
    };

    const int nk = K >> 6;                   // K / 64
    stage(0, 0);

    for (int it = 0; it < nk; ++it) {
        const int buf = it & 1;
        wait_async0();                        // my fills for `buf` are in LDS
        __syncthreads();                      // everyone's fills visible
        if (it + 1 < nk) stage(buf ^ 1, (it + 1) << 6);  // overlap with compute

        #pragma unroll
        for (int kk = 0; kk < 64; kk += 32) {
            // A fragments: 16x32; lanes<16 hold K 0-7 & 16-23 of the chunk,
            // lanes>=16 hold K 8-15 & 24-31 (ISA 05_wmma 16-bit A layout).
            Frag a[2];
            #pragma unroll
            for (int mi = 0; mi < 2; ++mi) {
                int row = wm * 32 + mi * 16 + l16;
                const unsigned short* p = &As[buf][row * 64 + kk + halfsel * 8];
                a[mi].u[0] = *(const uint4*)p;
                a[mi].u[1] = *(const uint4*)(p + 16);
            }
            // B fragments: 32x16; lanes<16 K 0-7 & 16-23, lanes>=16 K 8-15 & 24-31.
            #pragma unroll
            for (int ni = 0; ni < 4; ++ni) {
                Frag b;
                int col = wn * 64 + ni * 16 + l16;
                const unsigned short* p = &Bs[buf][col * 64 + kk + halfsel * 8];
                b.u[0] = *(const uint4*)p;
                b.u[1] = *(const uint4*)(p + 16);
                #pragma unroll
                for (int mi = 0; mi < 2; ++mi) {
                    acc[mi][ni] = __builtin_amdgcn_wmma_f32_16x16x32_bf16(
                        false, a[mi].v, false, b.v, (short)0, acc[mi][ni],
                        false, false);
                }
            }
        }
        __syncthreads();                      // done reading `buf`
    }

    // Epilogue. C layout: VGPR r holds row r (lanes 0-15) / row 8+r (16-31),
    // column = l16.
    #pragma unroll
    for (int ni = 0; ni < 4; ++ni) {
        int col = n0 + wn * 64 + ni * 16 + l16;
        float bv = bias ? bias[col] : 0.0f;
        #pragma unroll
        for (int mi = 0; mi < 2; ++mi) {
            int rbase = m0 + wm * 32 + mi * 16 + halfsel * 8;
            #pragma unroll
            for (int r = 0; r < 8; ++r) {
                float v = (acc[mi][ni][r] + bv) * scale;
                int row = rbase + r;
                if (OUTMODE == 0)
                    ((float*)Cout)[(size_t)row * ldc + col] = v;
                else if (OUTMODE == 1)
                    ((unsigned short*)Cout)[(size_t)row * ldc + col] = f32_to_bf16(v);
                else
                    ((unsigned short*)Cout)[(size_t)col * ldc + row] = f32_to_bf16(v);
            }
        }
    }
}

// ---------------------------------------------------------------- softmax
// One 256-thread block per row of S (4096 fp32). Writes bf16 P in place
// (row i of P at byte offset i*16KB, i.e. ldP = 8192 bf16 elements).
// All global reads happen before the first __syncthreads(), all writes after
// the last one, so the in-place overwrite is race-free.
__global__ void __launch_bounds__(256) softmax_rows(const float* __restrict__ S,
                                                    unsigned short* __restrict__ P,
                                                    int ldS, int ldP)
{
    __shared__ float red[8];
    const int row = blockIdx.x;
    const int tid = threadIdx.x;
    const float* s = S + (size_t)row * ldS;
    unsigned short* p = P + (size_t)row * ldP;

    float v[16];
    #pragma unroll
    for (int c = 0; c < 4; ++c) {
        float4 f = *(const float4*)&s[c * 1024 + tid * 4];
        v[c*4+0] = f.x; v[c*4+1] = f.y; v[c*4+2] = f.z; v[c*4+3] = f.w;
    }

    float mx = -3.4e38f;
    #pragma unroll
    for (int i = 0; i < 16; ++i) mx = fmaxf(mx, v[i]);
    for (int off = 16; off; off >>= 1) mx = fmaxf(mx, __shfl_xor(mx, off, 32));
    if ((tid & 31) == 0) red[tid >> 5] = mx;
    __syncthreads();
    if (tid < 32) {
        float t = (tid < 8) ? red[tid] : -3.4e38f;
        for (int off = 4; off; off >>= 1) t = fmaxf(t, __shfl_xor(t, off, 32));
        if (tid == 0) red[0] = t;
    }
    __syncthreads();
    mx = red[0];
    __syncthreads();

    float sum = 0.0f;
    #pragma unroll
    for (int i = 0; i < 16; ++i) { v[i] = __expf(v[i] - mx); sum += v[i]; }
    for (int off = 16; off; off >>= 1) sum += __shfl_xor(sum, off, 32);
    if ((tid & 31) == 0) red[tid >> 5] = sum;
    __syncthreads();
    if (tid < 32) {
        float t = (tid < 8) ? red[tid] : 0.0f;
        for (int off = 4; off; off >>= 1) t += __shfl_xor(t, off, 32);
        if (tid == 0) red[0] = t;
    }
    __syncthreads();
    const float inv = 1.0f / red[0];

    #pragma unroll
    for (int c = 0; c < 4; ++c) {
        unsigned short h[4];
        #pragma unroll
        for (int j = 0; j < 4; ++j) h[j] = f32_to_bf16(v[c*4+j] * inv);
        *(uint2*)&p[c * 1024 + tid * 4] = *(const uint2*)h;
    }
}

// ---------------------------------------------------------------- launch
extern "C" void kernel_launch(void* const* d_in, const int* in_sizes, int n_in,
                              void* d_out, int out_size, void* d_ws, size_t ws_size,
                              hipStream_t stream)
{
    (void)in_sizes; (void)n_in; (void)out_size; (void)ws_size;
    const float* x  = (const float*)d_in[0];
    const float* Wq = (const float*)d_in[1];
    const float* bq = (const float*)d_in[2];
    const float* Wk = (const float*)d_in[3];
    const float* bk = (const float*)d_in[4];
    const float* Wv = (const float*)d_in[5];
    const float* bv = (const float*)d_in[6];

    char* ws = (char*)d_ws;
    const size_t MB = 1024 * 1024;
    unsigned short* xb  = (unsigned short*)(ws + 0);       // 8 MB  x bf16 [4096,1024]
    unsigned short* wqb = (unsigned short*)(ws + 8  * MB); // 2 MB
    unsigned short* wkb = (unsigned short*)(ws + 10 * MB); // 2 MB
    unsigned short* wvb = (unsigned short*)(ws + 12 * MB); // 2 MB
    unsigned short* Qb  = (unsigned short*)(ws + 14 * MB); // 8 MB  (pre-scaled by 1/sqrt(D))
    unsigned short* Kb  = (unsigned short*)(ws + 22 * MB); // 8 MB
    unsigned short* Vtb = (unsigned short*)(ws + 30 * MB); // 8 MB  V^T [1024,4096]
    float*          Sf  = (float*)(ws + 38 * MB);          // 64 MB scores fp32 [4096,4096]
    unsigned short* Pb  = (unsigned short*)(ws + 38 * MB); // bf16 P interleaved, ld 8192

    // 1) fp32 -> bf16
    cvt_f32_bf16<<<(NTOK * DIMD / 4 + 255) / 256, 256, 0, stream>>>(x,  xb,  NTOK * DIMD / 4);
    cvt_f32_bf16<<<(DIMD * DIMD / 4 + 255) / 256, 256, 0, stream>>>(Wq, wqb, DIMD * DIMD / 4);
    cvt_f32_bf16<<<(DIMD * DIMD / 4 + 255) / 256, 256, 0, stream>>>(Wk, wkb, DIMD * DIMD / 4);
    cvt_f32_bf16<<<(DIMD * DIMD / 4 + 255) / 256, 256, 0, stream>>>(Wv, wvb, DIMD * DIMD / 4);

    const dim3 blk(256);
    const dim3 gProj(DIMD / 128, NTOK / 128);   // (8, 32)
    const dim3 gScor(NTOK / 128, NTOK / 128);   // (32, 32)
    const float qscale = 0.03125f;              // 1/sqrt(1024), folded into Q

    // 2) Q = (x @ Wq^T + bq) * 1/sqrt(D)   -> bf16 [4096,1024]
    wmma_gemm_abT<1><<<gProj, blk, 0, stream>>>(xb, wqb, bq, qscale, Qb, DIMD,
                                                NTOK, DIMD, DIMD, DIMD, DIMD);
    // 3) K = x @ Wk^T + bk                 -> bf16 [4096,1024]
    wmma_gemm_abT<1><<<gProj, blk, 0, stream>>>(xb, wkb, bk, 1.0f, Kb, DIMD,
                                                NTOK, DIMD, DIMD, DIMD, DIMD);
    // 4) V = x @ Wv^T + bv, stored transposed -> bf16 V^T [1024,4096]
    wmma_gemm_abT<2><<<gProj, blk, 0, stream>>>(xb, wvb, bv, 1.0f, Vtb, NTOK,
                                                NTOK, DIMD, DIMD, DIMD, DIMD);
    // 5) S = Q @ K^T  -> fp32 [4096,4096] (L2-resident)
    wmma_gemm_abT<0><<<gScor, blk, 0, stream>>>(Qb, Kb, nullptr, 1.0f, Sf, NTOK,
                                                NTOK, NTOK, DIMD, DIMD, DIMD);
    // 6) P = softmax(S) rows -> bf16 in place (ldP = 8192)
    softmax_rows<<<NTOK, 256, 0, stream>>>(Sf, Pb, NTOK, 2 * NTOK);
    // 7) out = P @ V = P @ (V^T)^T  -> fp32 [4096,1024]
    wmma_gemm_abT<0><<<gProj, blk, 0, stream>>>(Pb, Vtb, nullptr, 1.0f,
                                                (float*)d_out, DIMD,
                                                NTOK, DIMD, NTOK, 2 * NTOK, NTOK);
}
